// GCN_15298673508536
// MI455X (gfx1250) — matrix operand order repaired and checked by reading, not compiled, not run on hardware
//
#include <hip/hip_runtime.h>
#include <hip/hip_bf16.h>

#define NN 100000   // nodes
#define NE 600000   // edges
#define NG 512      // graphs
#define FD 128      // feature dim (all layers)
#define NNF (NN * FD)

typedef __attribute__((ext_vector_type(16))) __bf16 v16bf;
typedef __attribute__((ext_vector_type(8)))  __bf16 v8bf;
typedef __attribute__((ext_vector_type(4)))  __bf16 v4bf;
typedef __attribute__((ext_vector_type(8)))  float  v8f;

// ---------------- utility kernels ----------------

__global__ __launch_bounds__(256) void k_init(float* __restrict__ deg,
                                              float* __restrict__ pooled,
                                              float* __restrict__ counts) {
  int i = blockIdx.x * 256 + threadIdx.x;
  if (i < NN) deg[i] = 1.0f;            // self-loop contribution to degree
  if (i < NG * FD) pooled[i] = 0.0f;
  if (i < NG) counts[i] = 0.0f;
}

__global__ __launch_bounds__(256) void k_deg(const long long* __restrict__ dst,
                                             float* __restrict__ deg) {
  int e = blockIdx.x * 256 + threadIdx.x;
  if (e < NE) atomicAdd(&deg[(int)dst[e]], 1.0f);
}

__global__ __launch_bounds__(256) void k_rsqrt(float* __restrict__ deg) {
  int i = blockIdx.x * 256 + threadIdx.x;
  if (i < NN) deg[i] = rsqrtf(deg[i]);  // deg >= 1 always (self-loop)
}

// fp32 -> bf16 row-major copy (GEMM A operand), coalesced float4 -> 8B stores
__global__ __launch_bounds__(256) void k_cvt(const float* __restrict__ X,
                                             __bf16* __restrict__ Xb) {
  int tid = blockIdx.x * 256 + threadIdx.x;
  if (tid >= NNF / 4) return;
  const float4 v = *(const float4*)(X + (size_t)tid * 4);
  v4bf o;
  o[0] = (__bf16)v.x; o[1] = (__bf16)v.y; o[2] = (__bf16)v.z; o[3] = (__bf16)v.w;
  *(v4bf*)(Xb + (size_t)tid * 4) = o;
}

// ---------------- W pre-swizzle into per-lane WMMA-B fragment order ----------------
// Layout: chunk index ((kb*8+nt)*32 + lane)*16 + j holds W[kb*32 + 8*(lane>>4) + (j&7) + 16*(j>>3)][nt*16 + (lane&15)]
// so a lane's 16-element B fragment is one contiguous 32-byte run.
__global__ __launch_bounds__(256) void k_prepW(const float* __restrict__ Wm,
                                               __bf16* __restrict__ Wswz) {
  const int g = blockIdx.x * 256 + threadIdx.x;   // one 8-element chunk-half each
  if (g >= (FD * FD) / 8) return;
  const int jh   = g & 1;
  const int lane = (g >> 1) & 31;
  const int grp  = g >> 6;                         // kb*8 + nt
  const int kb = grp >> 3, nt = grp & 7;
  const int half = lane >> 4, lm = lane & 15;
  const int kbase = kb * 32 + half * 8 + jh * 16;
  const int n = nt * 16 + lm;
  v8bf w8;
#pragma unroll
  for (int j = 0; j < 8; ++j) w8[j] = (__bf16)Wm[(kbase + j) * FD + n];
  *(v8bf*)(Wswz + g * 8) = w8;                     // coalesced 16B store
}

// ---------------- WMMA GEMM: Y[NN,128] = Xb[NN,128](bf16) @ W[128,128] ----------------
// 256 threads = 8 wave32s; each wave computes a 16x128 strip.
// A: two global_load_b128 per K-block straight from row-major bf16 rows.
// B: two ds_load_b128 per fragment from the pre-swizzled LDS table.
__global__ __launch_bounds__(256) void k_gemm(const __bf16* __restrict__ Xb,
                                              const __bf16* __restrict__ Wswz,
                                              float* __restrict__ Y) {
  __shared__ __align__(16) __bf16 Wlds[FD * FD];   // 32 KB, swizzled fragment order

  const int t = threadIdx.x;
  const int wave = t >> 5, lane = t & 31;

  // coalesced wide copy of pre-swizzled W into LDS (b128 in / b128 out)
  for (int i = t; i < (FD * FD) / 8; i += 256)
    *(v8bf*)&Wlds[i * 8] = *(const v8bf*)(Wswz + i * 8);
  __syncthreads();

  const int half = lane >> 4, lm = lane & 15;
  const int rowBase = blockIdx.x * 128 + wave * 16;
  const int arow = rowBase + lm;
  const int ar = (arow < NN) ? arow : 0;           // clamp: junk rows are never stored
  const __bf16* xrow = Xb + (size_t)ar * FD;

  v8f acc[8] = {};
#pragma unroll
  for (int kb = 0; kb < 4; ++kb) {
    // A fragment: two contiguous 8-element K-runs per lane
    const v8bf alo = *(const v8bf*)(xrow + kb * 32 + half * 8);
    const v8bf ahi = *(const v8bf*)(xrow + kb * 32 + 16 + half * 8);
    const v16bf a = __builtin_shufflevector(alo, ahi,
        0, 1, 2, 3, 4, 5, 6, 7, 8, 9, 10, 11, 12, 13, 14, 15);
#pragma unroll
    for (int nt = 0; nt < 8; ++nt) {
      const __bf16* cp = &Wlds[((kb * 8 + nt) * 32 + lane) * 16];
      const v8bf blo = *(const v8bf*)cp;
      const v8bf bhi = *(const v8bf*)(cp + 8);
      const v16bf b = __builtin_shufflevector(blo, bhi,
          0, 1, 2, 3, 4, 5, 6, 7, 8, 9, 10, 11, 12, 13, 14, 15);
      acc[nt] = __builtin_amdgcn_wmma_f32_16x16x32_bf16(
          false, a, false, b, (short)0, acc[nt], false, false);
    }
  }

  // C/D layout: VGPR r holds M = r + 8*half, N = lm
#pragma unroll
  for (int nt = 0; nt < 8; ++nt) {
#pragma unroll
    for (int r = 0; r < 8; ++r) {
      const int row = rowBase + r + half * 8;
      if (row < NN) Y[(size_t)row * FD + nt * 16 + lm] = acc[nt][r];
    }
  }
}

// ---------------- aggregation ----------------

// agg[i,:] = dis[i]^2 * hW[i,:]   (self-loop term, avoids atomics for it)
__global__ __launch_bounds__(256) void k_selfinit(const float* __restrict__ dis,
                                                  const float* __restrict__ hW,
                                                  float* __restrict__ agg) {
  int tid = blockIdx.x * 256 + threadIdx.x;
  if (tid >= NNF) return;
  const int i = tid >> 7;
  const float d = dis[i];
  agg[tid] = d * d * hW[tid];
}

// one wave per edge: agg[dst,:] += dis[src]*dis[dst]*hW[src,:]
__global__ __launch_bounds__(256) void k_edges(const long long* __restrict__ src,
                                               const long long* __restrict__ dst,
                                               const float* __restrict__ dis,
                                               const float* __restrict__ hW,
                                               float* __restrict__ agg) {
  int tid = blockIdx.x * 256 + threadIdx.x;
  const int e = tid >> 5;
  if (e >= NE) return;
  const int q = tid & 31;
  const int s = (int)src[e];
  const int d = (int)dst[e];
  // gfx1250 prefetch of a future source row (working set is L2-resident)
  if (q == 0) {
    const int e2 = e + 4096;
    if (e2 < NE) __builtin_prefetch(&hW[(size_t)((int)src[e2]) * FD], 0, 0);
  }
  const float w = dis[s] * dis[d];
  const float4 v = *(const float4*)(hW + (size_t)s * FD + q * 4);
  float* a = agg + (size_t)d * FD + q * 4;
  atomicAdd(a + 0, w * v.x);
  atomicAdd(a + 1, w * v.y);
  atomicAdd(a + 2, w * v.z);
  atomicAdd(a + 3, w * v.w);
}

// layer-1 epilogue: h1 = relu(agg + b1), emitted directly as bf16 GEMM-A operand
__global__ __launch_bounds__(256) void k_bias_relu_cvt(const float* __restrict__ agg,
                                                       const float* __restrict__ b,
                                                       __bf16* __restrict__ Xb) {
  int tid = blockIdx.x * 256 + threadIdx.x;
  if (tid >= NNF / 4) return;
  const float4 v = *(const float4*)(agg + (size_t)tid * 4);
  const int fb = (tid * 4) & (FD - 1);
  float a0 = fmaxf(v.x + b[fb + 0], 0.0f);
  float a1 = fmaxf(v.y + b[fb + 1], 0.0f);
  float a2 = fmaxf(v.z + b[fb + 2], 0.0f);
  float a3 = fmaxf(v.w + b[fb + 3], 0.0f);
  v4bf o;
  o[0] = (__bf16)a0; o[1] = (__bf16)a1; o[2] = (__bf16)a2; o[3] = (__bf16)a3;
  *(v4bf*)(Xb + (size_t)tid * 4) = o;
}

// layer-2 epilogue: h2 = agg + b2 (f32, in place)
__global__ __launch_bounds__(256) void k_bias2(float* __restrict__ agg,
                                               const float* __restrict__ b) {
  int tid = blockIdx.x * 256 + threadIdx.x;
  if (tid >= NNF / 4) return;
  float4 v = *(const float4*)(agg + (size_t)tid * 4);
  const int fb = (tid * 4) & (FD - 1);
  v.x += b[fb + 0]; v.y += b[fb + 1]; v.z += b[fb + 2]; v.w += b[fb + 3];
  *(float4*)(agg + (size_t)tid * 4) = v;
}

// ---------------- pooling + head ----------------

__global__ __launch_bounds__(256) void k_pool(const long long* __restrict__ batch,
                                              const float* __restrict__ h,
                                              float* __restrict__ pooled,
                                              float* __restrict__ counts) {
  int tid = blockIdx.x * 256 + threadIdx.x;
  const int i = tid >> 5;
  if (i >= NN) return;
  const int q = tid & 31;
  const int g = (int)batch[i];
  const float4 v = *(const float4*)(h + (size_t)i * FD + q * 4);
  float* p = pooled + (size_t)g * FD + q * 4;
  atomicAdd(p + 0, v.x);
  atomicAdd(p + 1, v.y);
  atomicAdd(p + 2, v.z);
  atomicAdd(p + 3, v.w);
  if (q == 0) atomicAdd(&counts[g], 1.0f);
}

__global__ __launch_bounds__(256) void k_final(const float* __restrict__ pooled,
                                               const float* __restrict__ counts,
                                               const float* __restrict__ Wlin,
                                               const float* __restrict__ blin,
                                               float* __restrict__ out) {
  int g = blockIdx.x * 256 + threadIdx.x;
  if (g >= NG) return;
  float c = counts[g];
  c = c > 1.0f ? c : 1.0f;
  float s = 0.0f;
#pragma unroll 4
  for (int f = 0; f < FD; ++f) s += pooled[g * FD + f] * Wlin[f];
  out[g] = s / c + blin[0];
}

// ---------------- host-side orchestration ----------------

extern "C" void kernel_launch(void* const* d_in, const int* in_sizes, int n_in,
                              void* d_out, int out_size, void* d_ws, size_t ws_size,
                              hipStream_t stream) {
  const float*     x     = (const float*)d_in[0];
  const long long* ei    = (const long long*)d_in[1];   // [2, NE] flat
  const long long* batch = (const long long*)d_in[2];
  const float*     W1    = (const float*)d_in[3];
  const float*     b1    = (const float*)d_in[4];
  const float*     W2    = (const float*)d_in[5];
  const float*     b2    = (const float*)d_in[6];
  const float*     Wlin  = (const float*)d_in[7];
  const float*     blin  = (const float*)d_in[8];
  float* out = (float*)d_out;

  const long long* src = ei;
  const long long* dst = ei + NE;

  // workspace layout (float units, 16B-aligned sections)
  float*  ws     = (float*)d_ws;
  float*  dis    = ws;                          // NN (deg -> dis in place)
  float*  hW     = ws + 102400;                 // NNF
  float*  agg    = hW + (size_t)NNF;            // NNF (also h2 in place)
  float*  pooled = agg + (size_t)NNF;           // NG*FD
  float*  counts = pooled + NG * FD;            // NG
  float*  tail   = counts + NG;                 // (offset 25,768,448 floats: /4 -> 16B ok)
  __bf16* Xb     = (__bf16*)tail;               // NNF bf16 (GEMM A operand)
  __bf16* Wswz   = (__bf16*)(tail + NNF / 2);   // FD*FD bf16 (swizzled W)

  const int gN    = (NN + 255) / 256;
  const int gE    = (NE + 255) / 256;
  const int gNF   = (NNF + 255) / 256;
  const int gNF4  = (NNF / 4 + 255) / 256;
  const int gE32  = (NE * 32 + 255) / 256;
  const int gN32  = (NN * 32 + 255) / 256;
  const int gGemm = (NN + 127) / 128;
  const int gPrep = ((FD * FD) / 8 + 255) / 256;

  // degree / normalization
  k_init <<<gN, 256, 0, stream>>>(dis, pooled, counts);
  k_deg  <<<gE, 256, 0, stream>>>(dst, dis);
  k_rsqrt<<<gN, 256, 0, stream>>>(dis);

  // layer 1: hW = x @ W1 ; agg = D^-1/2 (A+I) D^-1/2 hW ; Xb = bf16(relu(agg + b1))
  k_cvt          <<<gNF4, 256, 0, stream>>>(x, Xb);
  k_prepW        <<<gPrep, 256, 0, stream>>>(W1, Wswz);
  k_gemm         <<<gGemm, 256, 0, stream>>>(Xb, Wswz, hW);
  k_selfinit     <<<gNF,  256, 0, stream>>>(dis, hW, agg);
  k_edges        <<<gE32, 256, 0, stream>>>(src, dst, dis, hW, agg);
  k_bias_relu_cvt<<<gNF4, 256, 0, stream>>>(agg, b1, Xb);

  // layer 2: hW = h1 @ W2 ; agg = norm-agg ; h2 = agg + b2 (f32, in place)
  k_prepW        <<<gPrep, 256, 0, stream>>>(W2, Wswz);
  k_gemm         <<<gGemm, 256, 0, stream>>>(Xb, Wswz, hW);
  k_selfinit     <<<gNF,  256, 0, stream>>>(dis, hW, agg);
  k_edges        <<<gE32, 256, 0, stream>>>(src, dst, dis, hW, agg);
  k_bias2        <<<gNF4, 256, 0, stream>>>(agg, b2);

  // global mean pool + linear head
  k_pool <<<gN32, 256, 0, stream>>>(batch, agg, pooled, counts);
  k_final<<<(NG + 255) / 256, 256, 0, stream>>>(pooled, counts, Wlin, blin, out);
}
